// Attention_9929964388721
// MI455X (gfx1250) — compile-verified, matrix operand order checked by reading, not run on hardware
//
#include <hip/hip_runtime.h>
#include <hip/hip_bf16.h>
#include <math.h>

#define B_    4
#define H_    4
#define N_    4096
#define D_    32
#define C_IN  256
#define HID   128   // HEADS*DIM_HEAD
#define SCALE_ 10.0f
#define EPS_   1e-12f

typedef __attribute__((ext_vector_type(16))) _Float16 v16h;
typedef __attribute__((ext_vector_type(8)))  _Float16 v8h;
typedef __attribute__((ext_vector_type(8)))  float    v8f;

// ---------------------------------------------------------------------------
// Kernel 1: QKV projection.  qkv[o][pos] = sum_c w_qkv[o][c] * x[b][c][pos]
// Writes q,k (f16, [b][h][n][32], unnormalized) and v transposed
// (f16, [b][h][32][n]) so attention fragment loads are contiguous 16B chunks.
// blockIdx.z splits the 384 output channels 4 ways for WGP utilization.
// ---------------------------------------------------------------------------
__global__ __launch_bounds__(256) void qkv_proj_kernel(
    const float* __restrict__ x, const float* __restrict__ w,
    _Float16* __restrict__ q, _Float16* __restrict__ k,
    _Float16* __restrict__ vT)
{
  __shared__ float xs[C_IN][64];
  const int b    = blockIdx.y;
  const int pos0 = blockIdx.x * 64;
  const int t    = threadIdx.x;

  for (int kk = 0; kk < 64; ++kk) {
    int e = t + kk * 256;
    int c = e >> 6, p = e & 63;
    xs[c][p] = x[((long)b * C_IN + c) * N_ + pos0 + p];
  }
  __syncthreads();

  const int p   = t & 63;
  const int olo = blockIdx.z * 4 + (t >> 6);   // o residue mod 16
  for (int o = olo; o < 3 * HID; o += 16) {
    const float* wr = w + (long)o * C_IN;
    float acc = 0.f;
    #pragma unroll 8
    for (int c = 0; c < C_IN; ++c) acc = fmaf(wr[c], xs[c][p], acc);

    const int which = o >> 7;          // 0=q 1=k 2=v
    const int rem   = o & 127;
    const int h     = rem >> 5;
    const int d     = rem & 31;
    const int pos   = pos0 + p;
    const _Float16 hv = (_Float16)acc;
    if (which == 0)
      q[((long)(b * H_ + h) * N_ + pos) * D_ + d] = hv;
    else if (which == 1)
      k[((long)(b * H_ + h) * N_ + pos) * D_ + d] = hv;
    else
      vT[((long)(b * H_ + h) * D_ + d) * N_ + pos] = hv;
  }
}

// ---------------------------------------------------------------------------
// Kernel 2: per-row L2 normalization of q and k (q,k contiguous in ws).
// ---------------------------------------------------------------------------
__global__ __launch_bounds__(256) void l2norm_kernel(_Float16* __restrict__ qk,
                                                     int rows)
{
  int r = blockIdx.x * 256 + threadIdx.x;
  if (r >= rows) return;
  _Float16* p = qk + (long)r * D_;
  float vals[D_];
  float ss = 0.f;
  #pragma unroll
  for (int i = 0; i < D_; ++i) {
    vals[i] = (float)p[i];
    ss = fmaf(vals[i], vals[i], ss);
  }
  float inv = 1.0f / fmaxf(sqrtf(ss), EPS_);
  #pragma unroll
  for (int i = 0; i < D_; ++i) p[i] = (_Float16)(vals[i] * inv);
}

// ---------------------------------------------------------------------------
// Fragment loader: 16x32 f16 tile in the CDNA5 WMMA A/B lane/K ordering.
// lane<16 : row/col = lane,    K elements {0..7, 16..23}
// lane>=16: row/col = lane-16, K elements {8..15, 24..31}
// ---------------------------------------------------------------------------
__device__ inline v16h load_16x32_frag(const _Float16* __restrict__ base,
                                       long rowStride, int lane)
{
  const int row = lane & 15;
  const int off = (lane >> 4) * 8;
  const _Float16* pr = base + (long)row * rowStride + off;
  v8h lo = *(const v8h*)(pr);
  v8h hi = *(const v8h*)(pr + 16);
  v16h r;
  #pragma unroll
  for (int i = 0; i < 8; ++i) { r[i] = lo[i]; r[8 + i] = hi[i]; }
  return r;
}

// ---------------------------------------------------------------------------
// Kernel 3: flash attention, transposed formulation.
//   S^T = (K tile) x Q^T   -> per-query stats live in lane columns:
//                             in-lane reduce + one xor-16 shuffle.
//   O^T += (V^T tile) x P^T, where P^T's B-fragment is *register-identical*
//   to S^T's C-fragment (pB[e]=f16(s0[e]), pB[8+e]=f16(s1[e])): no LDS,
//   no cross-lane transpose.  Epilogue: per-lane 1/l, coalesced stores.
// ---------------------------------------------------------------------------
__global__ __launch_bounds__(256) void flash_attn_kernel(
    const _Float16* __restrict__ q, const _Float16* __restrict__ k,
    const _Float16* __restrict__ vT, float* __restrict__ ao)
{
  const int lane = threadIdx.x & 31;
  const int wv   = threadIdx.x >> 5;
  const int b    = blockIdx.y >> 2;
  const int h    = blockIdx.y & 3;
  const long bh  = (long)b * H_ + h;
  const int  i0  = blockIdx.x * 128 + wv * 16;
  const int  col = lane & 15;

  // Q^T as B fragment (K-dim = d = 32, N = 16 queries), loaded once
  const v16h qB = load_16x32_frag(q + (bh * N_ + i0) * D_, D_, lane);

  v8f o0 = {}; v8f o1 = {};          // O^T accumulators: d rows 0-15 / 16-31
  float m = -INFINITY, l = 0.f;      // stats for query i0+col (replicated in halves)

  const _Float16* kbh = k  + bh * (long)N_ * D_;
  const _Float16* vbh = vT + bh * (long)D_ * N_;

  #pragma unroll 2
  for (int jt = 0; jt < N_; jt += 32) {
    // S^T tiles [j][i], j 0..15 and 16..31
    v16h ka0 = load_16x32_frag(kbh + (long)jt * D_,        D_, lane);
    v16h ka1 = load_16x32_frag(kbh + (long)(jt + 16) * D_, D_, lane);
    v8f z = {};
    v8f s0 = __builtin_amdgcn_wmma_f32_16x16x32_f16(false, ka0, false, qB,
                                                    (short)0, z, false, false);
    v8f s1 = __builtin_amdgcn_wmma_f32_16x16x32_f16(false, ka1, false, qB,
                                                    (short)0, z, false, false);

    // per-query max: in-lane tree over 16 j's, then merge halves (xor 16)
    float t = -INFINITY;
    #pragma unroll
    for (int r = 0; r < 8; ++r) {
      s0[r] *= SCALE_; s1[r] *= SCALE_;
      t = fmaxf(t, fmaxf(s0[r], s1[r]));
    }
    t = fmaxf(t, __shfl_xor(t, 16));
    const float mn = fmaxf(m, t);
    const float f  = __expf(m - mn);   // first iter: exp(-inf)=0
    m = mn;

    // exponentiate; P^T B-fragment == S^T C-fragment layout: direct convert
    float rs = 0.f;
    v16h pB;
    #pragma unroll
    for (int r = 0; r < 8; ++r) {
      s0[r] = __expf(s0[r] - mn);
      s1[r] = __expf(s1[r] - mn);
      rs += s0[r] + s1[r];
      pB[r]     = (_Float16)s0[r];
      pB[8 + r] = (_Float16)s1[r];
    }
    rs += __shfl_xor(rs, 16);
    l = l * f + rs;
    #pragma unroll
    for (int r = 0; r < 8; ++r) { o0[r] *= f; o1[r] *= f; }

    // O^T += V^T x P^T  (V^T A-fragments contiguous in j)
    v16h va0 = load_16x32_frag(vbh + jt,                 N_, lane);
    v16h va1 = load_16x32_frag(vbh + (long)16 * N_ + jt, N_, lane);
    o0 = __builtin_amdgcn_wmma_f32_16x16x32_f16(false, va0, false, pB,
                                                (short)0, o0, false, false);
    o1 = __builtin_amdgcn_wmma_f32_16x16x32_f16(false, va1, false, pB,
                                                (short)0, o1, false, false);
  }

  // epilogue: O^T / l, coalesced stores to ao[b][c=h*32+d][n]
  const float inv  = 1.0f / l;
  const int   hi8  = (lane >> 4) * 8;
  const int   cbase = h * 32;
  #pragma unroll
  for (int r = 0; r < 8; ++r) {
    const int d = r + hi8;
    ao[(long)(b * HID + cbase + d)      * N_ + i0 + col] = o0[r] * inv;
    ao[(long)(b * HID + cbase + 16 + d) * N_ + i0 + col] = o1[r] * inv;
  }
}

// ---------------------------------------------------------------------------
// Kernel 4: output projection.
// out[b][o][pos] = b_out[o] + sum_c w_out[o][c]*ao[b][c][pos]
// blockIdx.z splits the 256 output channels 4 ways.
// ---------------------------------------------------------------------------
__global__ __launch_bounds__(256) void out_proj_kernel(
    const float* __restrict__ ao, const float* __restrict__ w_out,
    const float* __restrict__ b_out, float* __restrict__ out)
{
  __shared__ float as_[HID][64];
  const int b    = blockIdx.y;
  const int pos0 = blockIdx.x * 64;
  const int t    = threadIdx.x;

  for (int kk = 0; kk < 32; ++kk) {
    int e = t + kk * 256;
    int c = e >> 6, p = e & 63;
    as_[c][p] = ao[((long)b * HID + c) * N_ + pos0 + p];
  }
  __syncthreads();

  const int p   = t & 63;
  const int olo = blockIdx.z * 4 + (t >> 6);
  for (int o = olo; o < C_IN; o += 16) {
    const float* wr = w_out + (long)o * HID;
    float acc = b_out[o];
    #pragma unroll 8
    for (int c = 0; c < HID; ++c) acc = fmaf(wr[c], as_[c][p], acc);
    out[((long)b * C_IN + o) * N_ + pos0 + p] = acc;
  }
}

// ---------------------------------------------------------------------------
extern "C" void kernel_launch(void* const* d_in, const int* in_sizes, int n_in,
                              void* d_out, int out_size, void* d_ws, size_t ws_size,
                              hipStream_t stream)
{
  const float* x     = (const float*)d_in[0];
  const float* w_qkv = (const float*)d_in[1];
  const float* w_out = (const float*)d_in[2];
  const float* b_out = (const float*)d_in[3];
  float*       out   = (float*)d_out;

  // workspace layout: q(4MiB f16) | k(4MiB f16) | vT(4MiB f16) | ao(8MiB f32)
  char* ws = (char*)d_ws;
  _Float16* q  = (_Float16*)(ws);
  _Float16* k  = (_Float16*)(ws + ((size_t)4  << 20));
  _Float16* vT = (_Float16*)(ws + ((size_t)8  << 20));
  float*    ao = (float*)   (ws + ((size_t)12 << 20));

  dim3 g1(N_ / 64, B_, 4);
  qkv_proj_kernel<<<g1, 256, 0, stream>>>(x, w_qkv, q, k, vT);

  int rows = 2 * B_ * H_ * N_;   // q + k rows (contiguous)
  l2norm_kernel<<<rows / 256, 256, 0, stream>>>(q, rows);

  dim3 g2(N_ / 128, B_ * H_);
  flash_attn_kernel<<<g2, 256, 0, stream>>>(q, k, vT, ao);

  dim3 g3(N_ / 64, B_, 4);
  out_proj_kernel<<<g3, 256, 0, stream>>>(ao, w_out, b_out, out);
}
